// QLSTM_65481071397986
// MI455X (gfx1250) — matrix operand order repaired and checked by reading, not compiled, not run on hardware
//
#include <hip/hip_runtime.h>
#include <hip/hip_bf16.h>
#include <math.h>

#define SEQ    256
#define BATCH  1024
#define IN_DIM 512
#define HID    256
#define COMB   768      // IN_DIM + HID
#define NTOT   1024     // 4 gates * HID

#define KCHUNK 64
#define LDA    (KCHUNK + 8)   // padded LDS row (floats) to spread banks

typedef __attribute__((ext_vector_type(16))) __bf16 v16bf;
typedef __attribute__((ext_vector_type(8)))  float  v8f;

typedef __attribute__((__vector_size__(16))) int vi4;
typedef __attribute__((address_space(1))) vi4 GVI4;  // global int4
typedef __attribute__((address_space(3))) vi4 LVI4;  // LDS int4

// ---- gfx1250 async global->LDS path (guarded; sync fallback otherwise) -------------
#if defined(__has_builtin)
#  if __has_builtin(__builtin_amdgcn_global_load_async_to_lds_b128)
#    define HAVE_ASYNC_LDS 1
#  endif
#  if __has_builtin(__builtin_amdgcn_s_wait_asynccnt)
#    define ASYNC_WAIT() __builtin_amdgcn_s_wait_asynccnt(0)
#  endif
#endif
#ifndef HAVE_ASYNC_LDS
#  define HAVE_ASYNC_LDS 0
#endif
#ifndef ASYNC_WAIT
#  if HAVE_ASYNC_LDS
#    define ASYNC_WAIT() asm volatile("s_wait_asynccnt 0x0" ::: "memory")
#  else
#    define ASYNC_WAIT() ((void)0)
#  endif
#endif

// ---------------- WMMA fragment loaders (wave32, v_wmma_f32_16x16x32_bf16) ----------
// A fragment 16x32 (MxK) from LDS-resident row-major f32 tile (leading dim ld).
// Layout: lanes 0-15 -> M=lane, K=0..15; lanes 16-31 -> M=lane-16, K=16..31 (matches B).
__device__ __forceinline__ v16bf frag_a_lds(const float* __restrict__ A0, int ld) {
  const int lane = threadIdx.x & 31;
  const int m    = lane & 15;
  const int kh   = (lane >> 4) * 16;
  const float* r = A0 + m * ld + kh;
  v16bf out;
#pragma unroll
  for (int j = 0; j < 8; ++j) out[j]     = (__bf16)r[j];
#pragma unroll
  for (int j = 0; j < 8; ++j) out[8 + j] = (__bf16)r[8 + j];
  return out;
}

// B fragment 32x16 (KxN): B[k][n] = W[n][k], W row-major bf16 (leading dim ld).
// Lanes 0-15 -> N=lane, K=0..15; lanes 16-31 -> N=lane-16, K=16..31 (32B contiguous/lane).
__device__ __forceinline__ v16bf frag_b_wt(const __bf16* __restrict__ W, int ld) {
  const int lane = threadIdx.x & 31;
  const int n    = lane & 15;
  const int kh   = (lane >> 4) * 16;
  const __bf16* r = W + n * ld + kh;
  v16bf out;
#pragma unroll
  for (int j = 0; j < 16; ++j) out[j] = r[j];
  return out;
}

// ---------------- Pack weights/biases to bf16 workspace -----------------------------
__global__ __launch_bounds__(256) void qlstm_pack(
    const float* __restrict__ Wf, const float* __restrict__ Wi,
    const float* __restrict__ Wu, const float* __restrict__ Wo,
    const float* __restrict__ bf_, const float* __restrict__ bi_,
    const float* __restrict__ bu_, const float* __restrict__ bo_,
    __bf16* __restrict__ Wb, float* __restrict__ biasP) {
  int idx = blockIdx.x * blockDim.x + threadIdx.x;
  if (idx < 4 * HID * COMB) {
    int g   = idx / (HID * COMB);
    int rem = idx - g * (HID * COMB);
    const float* W = (g == 0) ? Wf : (g == 1) ? Wi : (g == 2) ? Wu : Wo;
    Wb[idx] = (__bf16)W[rem];
  }
  if (idx < NTOT) {
    int g = idx >> 8, h = idx & 255;
    const float* b = (g == 0) ? bf_ : (g == 1) ? bi_ : (g == 2) ? bu_ : bo_;
    biasP[idx] = b[h];
  }
}

__global__ __launch_bounds__(256) void qlstm_zero(float* __restrict__ p, int n) {
  int i = blockIdx.x * blockDim.x + threadIdx.x;
  if (i < n) p[i] = 0.0f;
}

// ------- Shared GEMM skeleton: block = 64 rows x 256 cols, 8 waves ------------------
// Each wave: 16x128 strip = 8 wmma accumulators reusing one A fragment per K-step.
// A tile double-buffered in LDS via async global->LDS copies (ASYNCcnt).

__device__ __forceinline__ void fill_tile_async(
    float* __restrict__ dst, const float* __restrict__ src, int src_ld) {
  // 64 rows x KCHUNK floats = 1024 float4; 256 threads -> 4 each.
#pragma unroll
  for (int it = 0; it < 4; ++it) {
    const int i    = threadIdx.x + it * 256;
    const int row  = i >> 4;          // 16 float4 per row
    const int col4 = i & 15;
    const float* g = src + (size_t)row * src_ld + col4 * 4;
    float* d       = dst + row * LDA + col4 * 4;
#if HAVE_ASYNC_LDS
    __builtin_amdgcn_global_load_async_to_lds_b128((GVI4*)g, (LVI4*)d, 0, 0);
#else
    float4 v = *(const float4*)g;
    d[0] = v.x; d[1] = v.y; d[2] = v.z; d[3] = v.w;
#endif
  }
}

// ---------------- X projection: [SEQ*B, IN] @ [IN, 4H] + bias -> P (f32) ------------
__global__ __launch_bounds__(256) void qlstm_xproj(
    const float* __restrict__ X, const __bf16* __restrict__ Wb,
    const float* __restrict__ biasP, float* __restrict__ P) {
  __shared__ float Atile[2][64 * LDA];
  const int mBase = blockIdx.x * 64;
  const int nBase = blockIdx.y * 256;
  const int wave  = threadIdx.x >> 5;
  const int wm    = wave >> 1;            // 0..3: 16-row slice
  const int wn    = wave & 1;             // 0..1: 128-col slice
  const int n0    = nBase + wn * 128;
  const float* Xblk = X + (size_t)mBase * IN_DIM;

  v8f z = {0.f, 0.f, 0.f, 0.f, 0.f, 0.f, 0.f, 0.f};
  v8f acc[8] = {z, z, z, z, z, z, z, z};

  fill_tile_async(Atile[0], Xblk, IN_DIM);
  ASYNC_WAIT();
  __syncthreads();

  int buf = 0;
  for (int kc = 0; kc < IN_DIM; kc += KCHUNK) {
    if (kc + KCHUNK < IN_DIM)
      fill_tile_async(Atile[buf ^ 1], Xblk + kc + KCHUNK, IN_DIM);
    const float* A0 = &Atile[buf][(wm * 16) * LDA];
#pragma unroll
    for (int ks = 0; ks < KCHUNK; ks += 32) {
      v16bf a = frag_a_lds(A0 + ks, LDA);
#pragma unroll
      for (int j = 0; j < 8; ++j) {
        v16bf b = frag_b_wt(Wb + (size_t)(n0 + j * 16) * COMB + kc + ks, COMB);
        acc[j] = __builtin_amdgcn_wmma_f32_16x16x32_bf16(
            false, a, false, b, (short)0, acc[j], false, false);
      }
    }
    ASYNC_WAIT();
    __syncthreads();
    buf ^= 1;
  }

  const int lane = threadIdx.x & 31;
  const int nl   = lane & 15;
  const int mh   = (lane >> 4) * 8;
  const int m0   = mBase + wm * 16;
#pragma unroll
  for (int j = 0; j < 8; ++j) {
    const int n    = n0 + j * 16 + nl;
    const float bv = biasP[n];
#pragma unroll
    for (int r = 0; r < 8; ++r)
      P[(size_t)(m0 + mh + r) * NTOT + n] = acc[j][r] + bv;
  }
}

// ---------------- Per-step H projection: h_{t-1} @ Wh^T + P_t -> Gpre ---------------
__global__ __launch_bounds__(256) void qlstm_hproj(
    const float* __restrict__ Hprev, const __bf16* __restrict__ Wb,
    const float* __restrict__ Pt, float* __restrict__ Gpre) {
  __shared__ float Atile[2][64 * LDA];
  const int mBase = blockIdx.x * 64;
  const int nBase = blockIdx.y * 256;
  const int wave  = threadIdx.x >> 5;
  const int wm    = wave >> 1;
  const int wn    = wave & 1;
  const int n0    = nBase + wn * 128;
  const float* Hblk = Hprev + (size_t)mBase * HID;

  v8f z = {0.f, 0.f, 0.f, 0.f, 0.f, 0.f, 0.f, 0.f};
  v8f acc[8] = {z, z, z, z, z, z, z, z};

  fill_tile_async(Atile[0], Hblk, HID);
  ASYNC_WAIT();
  __syncthreads();

  int buf = 0;
  for (int kc = 0; kc < HID; kc += KCHUNK) {
    if (kc + KCHUNK < HID)
      fill_tile_async(Atile[buf ^ 1], Hblk + kc + KCHUNK, HID);
    const float* A0 = &Atile[buf][(wm * 16) * LDA];
#pragma unroll
    for (int ks = 0; ks < KCHUNK; ks += 32) {
      v16bf a = frag_a_lds(A0 + ks, LDA);
#pragma unroll
      for (int j = 0; j < 8; ++j) {
        v16bf b = frag_b_wt(Wb + (size_t)(n0 + j * 16) * COMB + IN_DIM + kc + ks, COMB);
        acc[j] = __builtin_amdgcn_wmma_f32_16x16x32_bf16(
            false, a, false, b, (short)0, acc[j], false, false);
      }
    }
    ASYNC_WAIT();
    __syncthreads();
    buf ^= 1;
  }

  const int lane = threadIdx.x & 31;
  const int nl   = lane & 15;
  const int mh   = (lane >> 4) * 8;
  const int m0   = mBase + wm * 16;
#pragma unroll
  for (int j = 0; j < 8; ++j) {
    const int n = n0 + j * 16 + nl;
#pragma unroll
    for (int r = 0; r < 8; ++r) {
      const size_t off = (size_t)(m0 + mh + r) * NTOT + n;
      Gpre[off] = acc[j][r] + Pt[off];
    }
  }
}

// ---------------- Per-step qlayer (cumprod of cos) + LSTM update --------------------
// 4 waves per block (one per gate), one batch row per block. Lane-chunked serial
// cumprod (8 elems) + 5-step __shfl_up wave scan; single barrier for gate exchange.
__global__ __launch_bounds__(128) void qlstm_step(
    const float* __restrict__ Gpre,
    const float* __restrict__ pf, const float* __restrict__ pi,
    const float* __restrict__ pu, const float* __restrict__ po,
    float* __restrict__ cbuf, float* __restrict__ hprev,
    float* __restrict__ outT) {
  __shared__ float sQ[4][HID];
  const int b    = blockIdx.x;
  const int g    = threadIdx.x >> 5;
  const int lane = threadIdx.x & 31;
  const float* p = (g == 0) ? pf : (g == 1) ? pi : (g == 2) ? pu : po;
  const float* G = Gpre + (size_t)b * NTOT + g * HID;
  const int h0   = lane * 8;

  float a[8];
  float run = 1.0f;
#pragma unroll
  for (int j = 0; j < 8; ++j) { run *= __cosf(G[h0 + j] + p[h0 + j]); a[j] = run; }

  float sc = run;                       // inclusive scan of chunk products
#pragma unroll
  for (int off = 1; off < 32; off <<= 1) {
    float v = __shfl_up(sc, off, 32);
    if (lane >= off) sc *= v;
  }
  float excl = __shfl_up(sc, 1, 32);    // exclusive prefix for this chunk
  if (lane == 0) excl = 1.0f;
#pragma unroll
  for (int j = 0; j < 8; ++j) sQ[g][h0 + j] = excl * a[j];
  __syncthreads();

#pragma unroll
  for (int r = 0; r < 2; ++r) {
    const int h = threadIdx.x + r * 128;
    const float f  = 1.f / (1.f + __expf(-sQ[0][h]));
    const float i  = 1.f / (1.f + __expf(-sQ[1][h]));
    const float g_ = tanhf(sQ[2][h]);
    const float o  = 1.f / (1.f + __expf(-sQ[3][h]));
    const size_t idx = (size_t)b * HID + h;
    const float c  = f * cbuf[idx] + i * g_;
    const float hv = o * tanhf(c);
    cbuf[idx]  = c;
    hprev[idx] = hv;
    outT[idx]  = hv;
  }
}

__global__ __launch_bounds__(256) void qlstm_finalize(
    const float* __restrict__ hprev, const float* __restrict__ cbuf,
    float* __restrict__ hx, float* __restrict__ cx) {
  int i = blockIdx.x * blockDim.x + threadIdx.x;
  if (i < BATCH * HID) { hx[i] = hprev[i]; cx[i] = cbuf[i]; }
}

// ---------------- Host orchestration ------------------------------------------------
extern "C" void kernel_launch(void* const* d_in, const int* in_sizes, int n_in,
                              void* d_out, int out_size, void* d_ws, size_t ws_size,
                              hipStream_t stream) {
  (void)in_sizes; (void)n_in; (void)out_size; (void)ws_size;
  const float* X   = (const float*)d_in[0];
  const float* Wf  = (const float*)d_in[1];
  const float* bf_ = (const float*)d_in[2];
  const float* Wi  = (const float*)d_in[3];
  const float* bi_ = (const float*)d_in[4];
  const float* Wu  = (const float*)d_in[5];
  const float* bu_ = (const float*)d_in[6];
  const float* Wo  = (const float*)d_in[7];
  const float* bo_ = (const float*)d_in[8];
  const float* pf  = (const float*)d_in[9];
  const float* pi  = (const float*)d_in[10];
  const float* pu  = (const float*)d_in[11];
  const float* po  = (const float*)d_in[12];

  char* ws = (char*)d_ws;
  size_t off = 0;
  __bf16* Wb   = (__bf16*)(ws + off); off += (size_t)4 * HID * COMB * sizeof(__bf16);
  float* biasP = (float*)(ws + off);  off += (size_t)NTOT * sizeof(float);
  off = (off + 255) & ~(size_t)255;
  float* hprev = (float*)(ws + off);  off += (size_t)BATCH * HID * sizeof(float);
  float* cbuf  = (float*)(ws + off);  off += (size_t)BATCH * HID * sizeof(float);
  float* Gpre  = (float*)(ws + off);  off += (size_t)BATCH * NTOT * sizeof(float);
  float* P     = (float*)(ws + off);  // SEQ*BATCH*NTOT f32 (~1 GB)

  float* outs = (float*)d_out;
  float* hx   = outs + (size_t)SEQ * BATCH * HID;
  float* cx   = hx + (size_t)BATCH * HID;

  qlstm_pack<<<(4 * HID * COMB + 255) / 256, 256, 0, stream>>>(
      Wf, Wi, Wu, Wo, bf_, bi_, bu_, bo_, Wb, biasP);
  qlstm_zero<<<(2 * BATCH * HID + 255) / 256, 256, 0, stream>>>(hprev, 2 * BATCH * HID);

  // One big fully-parallel GEMM for the input projections (2/3 of total FLOPs).
  qlstm_xproj<<<dim3(SEQ * BATCH / 64, NTOT / 256), 256, 0, stream>>>(X, Wb, biasP, P);

  // Sequential recurrence: WMMA h-GEMM + fused qlayer/LSTM update per step.
  for (int t = 0; t < SEQ; ++t) {
    qlstm_hproj<<<dim3(BATCH / 64, NTOT / 256), 256, 0, stream>>>(
        hprev, Wb, P + (size_t)t * BATCH * NTOT, Gpre);
    qlstm_step<<<BATCH, 128, 0, stream>>>(
        Gpre, pf, pi, pu, po, cbuf, hprev, outs + (size_t)t * BATCH * HID);
  }
  qlstm_finalize<<<(BATCH * HID + 255) / 256, 256, 0, stream>>>(hprev, cbuf, hx, cx);
}
